// FVTRStage_22711787062069
// MI455X (gfx1250) — matrix operand order repaired and checked by reading, not compiled, not run on hardware
//
#include <hip/hip_runtime.h>
#include <hip/hip_bf16.h>
#include <math.h>

// ---------------- problem constants ----------------
#define BSZ 64
#define CCH 256
#define HH  8
#define WW  64
#define LL  512           // W*H tokens
#define NHD 8
#define HD  32
#define NB  3
#define CO  512
#define HO  4
#define FF  1024
#define MROWS (BSZ*LL)    // 32768
#define MCONV (BSZ*HO*WW) // 16384
#define KCONV (3*CCH)     // 768

typedef __attribute__((ext_vector_type(16))) __bf16 v16bf;
typedef __attribute__((ext_vector_type(8)))  __bf16 v8bf;
typedef __attribute__((ext_vector_type(8)))  float  v8f;

// ---------------- WMMA helpers (CDNA5 wave32 layouts, ISA 7.12.2) ----------------
__device__ inline v8f wmma_bf16(v16bf a, v16bf b, v8f c) {
  return __builtin_amdgcn_wmma_f32_16x16x32_bf16(
      /*neg_a=*/false, a, /*neg_b=*/false, b,
      /*c_mod=*/(short)0, c, /*reuse_a=*/false, /*reuse_b=*/false);
}

// A-matrix 16x32 (row-major source, ld elements):
// lanes 0-15: row=lane, K = k0+{0..7, 16..23}; lanes 16-31: K = k0+{8..15, 24..31}
__device__ inline v16bf load_a_frag(const __bf16* p0, int ld, int row0, int k0, int lane) {
  int r  = row0 + (lane & 15);
  int kk = k0 + ((lane >> 4) << 3);
  const __bf16* p = p0 + (size_t)r * ld + kk;
  v8bf lo = *(const v8bf*)(p);
  v8bf hi = *(const v8bf*)(p + 16);
  return __builtin_shufflevector(lo, hi, 0,1,2,3,4,5,6,7,8,9,10,11,12,13,14,15);
}

// B-matrix 32x16 given TRANSPOSED storage BT [N x K] (row-major, ld elements):
// lanes 0-15: n=lane, K = k0+{0..7, 16..23} ; lanes 16-31: n=lane-16, K = k0+{8..15, 24..31}
// (same K order as the A fragment)
__device__ inline v16bf load_b_frag(const __bf16* p0, int ld, int col0, int k0, int lane) {
  int n  = col0 + (lane & 15);
  int kk = k0 + ((lane >> 4) << 3);
  const __bf16* p = p0 + (size_t)n * ld + kk;
  v8bf lo = *(const v8bf*)(p);
  v8bf hi = *(const v8bf*)(p + 16);
  return __builtin_shufflevector(lo, hi, 0,1,2,3,4,5,6,7,8,9,10,11,12,13,14,15);
}

// C/D 16x16 fp32: lane n=lane&15; VGPR r -> M = r + (lane>=16 ? 8 : 0)
__device__ inline void store_tile_bf16(__bf16* p0, int ld, int row0, int col0, v8f acc, int lane) {
  int n  = col0 + (lane & 15);
  int mb = row0 + ((lane >> 4) << 3);
  __bf16* p = p0 + (size_t)mb * ld + n;
#pragma unroll
  for (int r = 0; r < 8; ++r) p[(size_t)r * ld] = (__bf16)acc[r];
}

__device__ inline float gelu_exact(float x) {
  return 0.5f * x * (1.0f + erff(x * 0.70710678118654752f));
}

// local mask, reference-faithful: flat index interpreted h-major (l = h*W + w)
__device__ inline float mask_val(int l, int m) {
  int dh = (l >> 6) - (m >> 6); dh = dh < 0 ? -dh : dh;
  int dw = (l & 63) - (m & 63); dw = dw < 0 ? -dw : dw;
  return (dh <= 3 && dw <= 5) ? 1.0f : 0.0f;
}

// ---------------- weight prep ----------------
// dst[n*K+k] = src[k*N+n] as bf16; batch via blockIdx.y
__global__ void k_transpose_bf16(const float* __restrict__ src, __bf16* __restrict__ dst,
                                 int Kd, int Nd) {
  size_t total = (size_t)Kd * Nd;
  size_t i = (size_t)blockIdx.x * blockDim.x + threadIdx.x;
  if (i >= total) return;
  size_t base = (size_t)blockIdx.y * total;
  int n = (int)(i % Nd), k = (int)(i / Nd);
  dst[base + (size_t)n * Kd + k] = (__bf16)src[base + i];
}

// conv_w [CO, C, 3, 1] -> wct [CO][kh*C + ci]  (transposed GEMM-B layout)
__global__ void k_convw(const float* __restrict__ cw, __bf16* __restrict__ wct) {
  size_t i = (size_t)blockIdx.x * blockDim.x + threadIdx.x;
  if (i >= (size_t)CO * KCONV) return;
  int col = (int)(i % KCONV), co = (int)(i / KCONV);
  int kh = col / CCH, ci = col % CCH;
  wct[i] = (__bf16)cw[((size_t)co * CCH + ci) * 3 + kh];
}

// ---------------- layout kernels ----------------
// image [B,C,H,W] -> x [B, L=w*H+h, C] fp32
__global__ void k_permute_in(const float* __restrict__ img, float* __restrict__ x) {
  size_t i = (size_t)blockIdx.x * blockDim.x + threadIdx.x;  // index into x
  if (i >= (size_t)BSZ * LL * CCH) return;
  int c = (int)(i % CCH);
  size_t r = i / CCH;
  int t = (int)(r % LL);
  int b = (int)(r / LL);
  int w = t / HH, h = t % HH;
  x[i] = img[(((size_t)b * CCH + c) * HH + h) * WW + w];
}

// im2col for conv (3,1)/stride(2,1)/pad(1,0): Ac[(b*HO+ho)*W+w][kh*C+ci]
__global__ void k_im2col(const float* __restrict__ x, __bf16* __restrict__ Ac) {
  size_t i = (size_t)blockIdx.x * blockDim.x + threadIdx.x;
  if (i >= (size_t)MCONV * KCONV) return;
  int col = (int)(i % KCONV);
  size_t r = i / KCONV;
  int kh = col / CCH, ci = col % CCH;
  int wi = (int)(r % WW);
  int t2 = (int)(r / WW);
  int ho = t2 % HO, b = t2 / HO;
  int h = 2 * ho - 1 + kh;
  float v = 0.0f;
  if (h >= 0 && h < HH) v = x[((size_t)b * LL + (wi * HH + h)) * CCH + ci];
  Ac[i] = (__bf16)v;
}

// ---------------- layernorm (one wave per row) ----------------
__global__ __launch_bounds__(256)
void k_layernorm(const float* __restrict__ xin, const float* __restrict__ w,
                 const float* __restrict__ bb, float* __restrict__ outf,
                 __bf16* __restrict__ outb, int rows, int Cc) {
  int wave = threadIdx.x >> 5, lane = threadIdx.x & 31;
  int row = blockIdx.x * (blockDim.x >> 5) + wave;
  if (row >= rows) return;
  const float* xr = xin + (size_t)row * Cc;
  int per = Cc >> 5;
  float s = 0.f, s2 = 0.f;
  for (int i = 0; i < per; ++i) { float v = xr[lane + 32 * i]; s += v; s2 += v * v; }
#pragma unroll
  for (int off = 16; off; off >>= 1) { s += __shfl_xor(s, off, 32); s2 += __shfl_xor(s2, off, 32); }
  float mean = s / (float)Cc;
  float var  = s2 / (float)Cc - mean * mean;
  float rstd = rsqrtf(var + 1e-5f);
  for (int i = 0; i < per; ++i) {
    int c = lane + 32 * i;
    float v = (xr[c] - mean) * rstd * w[c] + bb[c];
    if (outf) outf[(size_t)row * Cc + c] = v;
    outb[(size_t)row * Cc + c] = (__bf16)v;
  }
}

// final merging-block LN over 512 channels, scatter to [B,CO,HO,W]
__global__ __launch_bounds__(256)
void k_final_ln(const float* __restrict__ yt, const float* __restrict__ w,
                const float* __restrict__ bb, float* __restrict__ out) {
  int wave = threadIdx.x >> 5, lane = threadIdx.x & 31;
  int row = blockIdx.x * 8 + wave;
  if (row >= MCONV) return;
  const float* xr = yt + (size_t)row * CO;
  float s = 0.f, s2 = 0.f;
#pragma unroll
  for (int i = 0; i < 16; ++i) { float v = xr[lane + 32 * i]; s += v; s2 += v * v; }
#pragma unroll
  for (int off = 16; off; off >>= 1) { s += __shfl_xor(s, off, 32); s2 += __shfl_xor(s2, off, 32); }
  float mean = s / (float)CO;
  float rstd = rsqrtf(s2 / (float)CO - mean * mean + 1e-5f);
  int wi = row % WW, t2 = row / WW;
  int ho = t2 % HO, b = t2 / HO;
#pragma unroll
  for (int i = 0; i < 16; ++i) {
    int c = lane + 32 * i;
    float v = (xr[c] - mean) * rstd * w[c] + bb[c];
    out[(((size_t)b * CO + c) * HO + ho) * WW + wi] = v;
  }
}

// ---------------- QKV projection (per b,h: [512x32] @ [32x32], K=32 == one WMMA) ----------------
__global__ __launch_bounds__(256)
void k_qkv(const __bf16* __restrict__ xnb,
           const __bf16* __restrict__ wqt, const __bf16* __restrict__ wkt,
           const __bf16* __restrict__ wvt,
           __bf16* __restrict__ qb, __bf16* __restrict__ kb, __bf16* __restrict__ vt) {
  int bh = blockIdx.x;
  int hh = bh % NHD;
  int wave = threadIdx.x >> 5, lane = threadIdx.x & 31;
  const __bf16* Ab = xnb + (size_t)(bh / NHD) * LL * CCH + hh * HD;  // lda = CCH
  for (int tid = wave; tid < 192; tid += 8) {   // 3 mats * 32 mtiles * 2 ntiles
    int mat = tid / 64, rem = tid % 64;
    int mt = rem >> 1, nt = rem & 1;
    const __bf16* Wt = (mat == 0 ? wqt : mat == 1 ? wkt : wvt) + (size_t)hh * HD * HD; // [E x D]
    v16bf a = load_a_frag(Ab, CCH, mt * 16, 0, lane);
    v16bf bf = load_b_frag(Wt, HD, nt * 16, 0, lane);
    v8f acc = {0.f,0.f,0.f,0.f,0.f,0.f,0.f,0.f};
    acc = wmma_bf16(a, bf, acc);
    if (mat == 2) {  // store V transposed: vt[bh][e][l]
      __bf16* vp = vt + (size_t)bh * HD * LL;
      int n  = nt * 16 + (lane & 15);
      int mb = mt * 16 + ((lane >> 4) << 3);
#pragma unroll
      for (int r = 0; r < 8; ++r) vp[(size_t)n * LL + mb + r] = (__bf16)acc[r];
    } else {
      __bf16* op = (mat == 0 ? qb : kb) + (size_t)bh * LL * HD;
      store_tile_bf16(op, HD, mt * 16, nt * 16, acc, lane);
    }
  }
}

// ---------------- attention: 32 query rows per block; S in LDS; softmax*mask; P@V ----------------
__global__ __launch_bounds__(256)
void k_attn(const __bf16* __restrict__ qb, const __bf16* __restrict__ kb,
            const __bf16* __restrict__ vt, __bf16* __restrict__ aob, int use_mask) {
  extern __shared__ char smem[];
  const int ldS = 516, ldP = 520;                 // pad vs 64 LDS banks
  float*  S = (float*)smem;                       // 32 x 516 fp32
  __bf16* P = (__bf16*)(smem + 32 * ldS * 4);     // 32 x 520 bf16
  int bh = blockIdx.x;
  int row0 = blockIdx.y * 32;
  int wave = threadIdx.x >> 5, lane = threadIdx.x & 31;
  const __bf16* Q  = qb + (size_t)bh * LL * HD;
  const __bf16* K  = kb + (size_t)bh * LL * HD;
  const __bf16* Vt = vt + (size_t)bh * HD * LL;
  const float scale = 0.17677669529663687f;       // 1/sqrt(32)

  // Phase 1: S = scale * Q_blk @ K^T  (2 x 32 tiles, K=hd=32 -> one WMMA each)
  for (int t = wave; t < 64; t += 8) {
    int mt = t >> 5, nt = t & 31;
    v16bf a  = load_a_frag(Q, HD, row0 + mt * 16, 0, lane);
    v16bf bf = load_b_frag(K, HD, nt * 16, 0, lane);
    v8f acc = {0.f,0.f,0.f,0.f,0.f,0.f,0.f,0.f};
    acc = wmma_bf16(a, bf, acc);
    int n  = nt * 16 + (lane & 15);
    int mb = mt * 16 + ((lane >> 4) << 3);
#pragma unroll
    for (int r = 0; r < 8; ++r) S[(size_t)(mb + r) * ldS + n] = acc[r] * scale;
  }
  __syncthreads();

  // Phase 2: rowwise stable softmax, then post-softmax multiplicative mask -> bf16 P
  for (int rr = 0; rr < 4; ++rr) {
    int lr = wave * 4 + rr;
    int gl = row0 + lr;
    float* Sr = &S[(size_t)lr * ldS];
    float mx = -1e30f;
    for (int i = lane; i < LL; i += 32) mx = fmaxf(mx, Sr[i]);
#pragma unroll
    for (int off = 16; off; off >>= 1) mx = fmaxf(mx, __shfl_xor(mx, off, 32));
    float sum = 0.f;
    for (int i = lane; i < LL; i += 32) { float e = __expf(Sr[i] - mx); Sr[i] = e; sum += e; }
#pragma unroll
    for (int off = 16; off; off >>= 1) sum += __shfl_xor(sum, off, 32);
    float inv = 1.0f / sum;
    for (int i = lane; i < LL; i += 32) {
      float mv = use_mask ? mask_val(gl, i) : 1.0f;
      P[(size_t)lr * ldP + i] = (__bf16)(Sr[i] * inv * mv);
    }
  }
  __syncthreads();

  // Phase 3: O = P[32x512] @ V[512x32], split-K over all 8 waves so every wave
  // reaches the WMMA with full EXEC (ISA: WMMA requires EXEC all-1s).
  // wave = (khalf<<2) | (mt<<1) | nt ; partials combined through LDS (S reuse).
  {
    int mt = (wave >> 1) & 1, nt = wave & 1, kh2 = wave >> 2;
    v8f acc = {0.f,0.f,0.f,0.f,0.f,0.f,0.f,0.f};
    int kbeg = kh2 * 256;
#pragma unroll
    for (int k0 = 0; k0 < 256; k0 += 32) {
      v16bf a  = load_a_frag(P, ldP, mt * 16, kbeg + k0, lane);
      v16bf bf = load_b_frag(Vt, LL, nt * 16, kbeg + k0, lane);
      acc = wmma_bf16(a, bf, acc);
    }
    float* part = S;  // 8 tiles x 256 floats = 8KB (S is free after phase 2)
    int n = lane & 15, off8 = (lane >> 4) << 3;
#pragma unroll
    for (int r = 0; r < 8; ++r) part[wave * 256 + (off8 + r) * 16 + n] = acc[r];
    __syncthreads();
    if (wave < 4) {
      int b = bh / NHD, hh = bh % NHD;
      __bf16* op = aob + (size_t)b * LL * CCH + hh * HD;  // [B,L,C] slice, ld=CCH
      int cmt = (wave >> 1) & 1, cnt = wave & 1;
#pragma unroll
      for (int r = 0; r < 8; ++r) {
        float v = part[wave * 256 + (off8 + r) * 16 + n] +
                  part[(wave + 4) * 256 + (off8 + r) * 16 + n];
        int gm = row0 + cmt * 16 + off8 + r;
        op[(size_t)gm * CCH + cnt * 16 + n] = (__bf16)v;
      }
    }
  }
}

// ---------------- generic WMMA GEMM with fused epilogue ----------------
// C = A[MxK]bf16 @ B (given transposed BT[NxK] bf16) (+bias) (gelu) (+res) -> f32 and/or bf16
// block = 8 waves; wave owns 32(M) x 64(N); block tile 128x128
template<int BIAS, int RES, int GELU_, int OUTF, int OUTB>
__global__ __launch_bounds__(256)
void k_gemm(const __bf16* __restrict__ A, const __bf16* __restrict__ BT,
            const float* __restrict__ bias, const float* __restrict__ res,
            float* __restrict__ Cf, __bf16* __restrict__ Cb,
            int M, int N, int Kd) {
  int wave = threadIdx.x >> 5, lane = threadIdx.x & 31;
  int row0 = blockIdx.y * 128 + (wave >> 1) * 32;
  int col0 = blockIdx.x * 128 + (wave & 1) * 64;
  v8f acc[2][4];
#pragma unroll
  for (int i = 0; i < 2; ++i)
#pragma unroll
    for (int j = 0; j < 4; ++j)
      acc[i][j] = (v8f){0.f,0.f,0.f,0.f,0.f,0.f,0.f,0.f};

  for (int k0 = 0; k0 < Kd; k0 += 32) {
    v16bf a0 = load_a_frag(A, Kd, row0,      k0, lane);
    v16bf a1 = load_a_frag(A, Kd, row0 + 16, k0, lane);
#pragma unroll
    for (int j = 0; j < 4; ++j) {
      v16bf bf = load_b_frag(BT, Kd, col0 + j * 16, k0, lane);
      acc[0][j] = wmma_bf16(a0, bf, acc[0][j]);
      acc[1][j] = wmma_bf16(a1, bf, acc[1][j]);
    }
  }

  int ln_  = lane & 15;
  int off8 = (lane >> 4) << 3;
#pragma unroll
  for (int i = 0; i < 2; ++i) {
#pragma unroll
    for (int j = 0; j < 4; ++j) {
      int n  = col0 + j * 16 + ln_;
      int mb = row0 + i * 16 + off8;
      float bv = BIAS ? bias[n] : 0.0f;
#pragma unroll
      for (int r = 0; r < 8; ++r) {
        float v = acc[i][j][r] + bv;
        if (GELU_) v = gelu_exact(v);
        if (RES)  v += res[(size_t)(mb + r) * N + n];
        if (OUTF) Cf[(size_t)(mb + r) * N + n] = v;
        if (OUTB) Cb[(size_t)(mb + r) * N + n] = (__bf16)v;
      }
    }
  }
}

// ---------------- host orchestration ----------------
extern "C" void kernel_launch(void* const* d_in, const int* in_sizes, int n_in,
                              void* d_out, int out_size, void* d_ws, size_t ws_size,
                              hipStream_t stream) {
  (void)in_sizes; (void)n_in; (void)out_size; (void)ws_size;
  const float* img    = (const float*)d_in[0];
  const float* ln1w   = (const float*)d_in[1];
  const float* ln1b   = (const float*)d_in[2];
  const float* ln2w   = (const float*)d_in[3];
  const float* ln2b   = (const float*)d_in[4];
  const float* wq     = (const float*)d_in[5];
  const float* wk     = (const float*)d_in[6];
  const float* wv     = (const float*)d_in[7];
  const float* wo     = (const float*)d_in[8];
  const float* w1     = (const float*)d_in[9];
  const float* b1     = (const float*)d_in[10];
  const float* w2     = (const float*)d_in[11];
  const float* b2     = (const float*)d_in[12];
  const float* convw  = (const float*)d_in[13];
  const float* convb  = (const float*)d_in[14];
  const float* mnw    = (const float*)d_in[15];
  const float* mnb    = (const float*)d_in[16];
  float* out = (float*)d_out;

  // workspace carve-up (all sizes are multiples of 256B)
  char* ws = (char*)d_ws;
  size_t off = 0;
  auto carve = [&](size_t bytes) { size_t o = off; off += (bytes + 255) & ~(size_t)255; return o; };
  float*  x    = (float*)(ws + carve((size_t)MROWS * CCH * 4));        // residual stream
  float*  xn   = (float*)(ws + carve((size_t)MROWS * CCH * 4));        // post-LN fp32
  __bf16* xnb  = (__bf16*)(ws + carve((size_t)MROWS * CCH * 2));       // post-LN bf16
  __bf16* qbuf = (__bf16*)(ws + carve((size_t)BSZ * NHD * LL * HD * 2));
  __bf16* kbuf = (__bf16*)(ws + carve((size_t)BSZ * NHD * LL * HD * 2));
  __bf16* vtb  = (__bf16*)(ws + carve((size_t)BSZ * NHD * HD * LL * 2));
  __bf16* aob  = (__bf16*)(ws + carve((size_t)MROWS * CCH * 2));
  __bf16* hb   = (__bf16*)(ws + carve((size_t)MROWS * FF * 2));
  __bf16* wqt  = (__bf16*)(ws + carve((size_t)NB * NHD * HD * HD * 2));
  __bf16* wkt  = (__bf16*)(ws + carve((size_t)NB * NHD * HD * HD * 2));
  __bf16* wvt  = (__bf16*)(ws + carve((size_t)NB * NHD * HD * HD * 2));
  __bf16* wot  = (__bf16*)(ws + carve((size_t)NB * CCH * CCH * 2));
  __bf16* w1t  = (__bf16*)(ws + carve((size_t)NB * CCH * FF * 2));
  __bf16* w2t  = (__bf16*)(ws + carve((size_t)NB * FF * CCH * 2));
  __bf16* wct  = (__bf16*)(ws + carve((size_t)CO * KCONV * 2));
  // late-stage aliases (q/k and h regions are dead by then)
  __bf16* Ac = qbuf;        // im2col: 16384*768*2 = 24MB <= q+k region (32MB)
  float*  yt = (float*)hb;  // conv output: 16384*512*4 = 32MB <= h region (64MB)

  // ---- weight prep (bf16 + transpose to [N x K]) ----
  k_transpose_bf16<<<dim3(4, NB * NHD), 256, 0, stream>>>(wq, wqt, HD, HD);
  k_transpose_bf16<<<dim3(4, NB * NHD), 256, 0, stream>>>(wk, wkt, HD, HD);
  k_transpose_bf16<<<dim3(4, NB * NHD), 256, 0, stream>>>(wv, wvt, HD, HD);
  k_transpose_bf16<<<dim3(256, NB), 256, 0, stream>>>(wo, wot, CCH, CCH);
  k_transpose_bf16<<<dim3(1024, NB), 256, 0, stream>>>(w1, w1t, CCH, FF);
  k_transpose_bf16<<<dim3(1024, NB), 256, 0, stream>>>(w2, w2t, FF, CCH);
  k_convw<<<(CO * KCONV + 255) / 256, 256, 0, stream>>>(convw, wct);

  // ---- ToFiber permute ----
  k_permute_in<<<(MROWS * CCH + 255) / 256, 256, 0, stream>>>(img, x);

  const int locals[NB] = {1, 1, 0};
  const size_t attn_lds = 32 * 516 * 4 + 32 * 520 * 2;  // 99328 B

  for (int i = 0; i < NB; ++i) {
    // LN1 -> xn (fp32 residual) + xnb (bf16 GEMM operand)
    k_layernorm<<<MROWS / 8, 256, 0, stream>>>(x, ln1w + i * CCH, ln1b + i * CCH,
                                               xn, xnb, MROWS, CCH);
    // QKV (V written transposed)
    k_qkv<<<BSZ * NHD, 256, 0, stream>>>(xnb,
        wqt + (size_t)i * NHD * HD * HD, wkt + (size_t)i * NHD * HD * HD,
        wvt + (size_t)i * NHD * HD * HD, qbuf, kbuf, vtb);
    // attention (softmax + post-softmax local mask) -> aob [B,L,C] bf16
    k_attn<<<dim3(BSZ * NHD, LL / 32), 256, attn_lds, stream>>>(qbuf, kbuf, vtb, aob, locals[i]);
    // x = aob @ Wo + xn   (residual is the LN1 output)
    k_gemm<0,1,0,1,0><<<dim3(CCH / 128, MROWS / 128), 256, 0, stream>>>(
        aob, wot + (size_t)i * CCH * CCH, nullptr, xn, x, nullptr, MROWS, CCH, CCH);
    // LN2
    k_layernorm<<<MROWS / 8, 256, 0, stream>>>(x, ln2w + i * CCH, ln2b + i * CCH,
                                               xn, xnb, MROWS, CCH);
    // h = gelu(xnb @ W1 + b1)  -> bf16
    k_gemm<1,0,1,0,1><<<dim3(FF / 128, MROWS / 128), 256, 0, stream>>>(
        xnb, w1t + (size_t)i * CCH * FF, b1 + i * FF, nullptr, nullptr, hb, MROWS, FF, CCH);
    // x = h @ W2 + b2 + xn   (residual is the LN2 output)
    k_gemm<1,1,0,1,0><<<dim3(CCH / 128, MROWS / 128), 256, 0, stream>>>(
        hb, w2t + (size_t)i * FF * CCH, b2 + i * CCH, xn, x, nullptr, MROWS, CCH, FF);
  }

  // ---- merging block: im2col + conv-GEMM + channel LN ----
  k_im2col<<<((size_t)MCONV * KCONV + 255) / 256, 256, 0, stream>>>(x, Ac);
  k_gemm<1,0,0,1,0><<<dim3(CO / 128, MCONV / 128), 256, 0, stream>>>(
      Ac, wct, convb, nullptr, yt, nullptr, MCONV, CO, KCONV);
  k_final_ln<<<MCONV / 8, 256, 0, stream>>>(yt, mnw, mnb, out);
}
